// NMPEUInteraction_44590350467105
// MI455X (gfx1250) — matrix operand order repaired and checked by reading, not compiled, not run on hardware
//
#include <hip/hip_runtime.h>

// ---------------------------------------------------------------------------
// Fused SchNet NMP-EU interaction for gfx1250 (MI455X).
// bf16 WMMA (v_wmma_f32_16x16x32_bf16) with f32 accumulate; weights staged to
// LDS as [K][pitch] dwords of bf16 N-pairs (odd pitch -> conflict-free);
// each edge wave owns TWO 16-row M-tiles so every B-fragment ds_load feeds two
// WMMAs; inter-GEMM C->A relayout via per-wave LDS staging + shfl pairing;
// segment_sum via global f32 atomics (agg fits in the 192MB L2).
// ---------------------------------------------------------------------------

#define DIMF 64
#define RBFF 64
#define KIN  192          // 2*dim + rbf
#define HID  128          // 2*dim
#define EW   4            // edge-kernel waves per block (2 M-tiles each)
#define NW   8            // node-kernel waves per block
#define P1 65             // LDS pitch (dwords) for N=128 weight matrix
#define P2 33             // LDS pitch (dwords) for N=64 weight matrices

typedef __attribute__((ext_vector_type(16))) __bf16 v16bf;
typedef __attribute__((ext_vector_type(8)))  float  v8f;

union Frag {
  v16bf   bf;
  unsigned u[8];
};

#define WMMA_BF16(a, b, c) \
  __builtin_amdgcn_wmma_f32_16x16x32_bf16(false, (a), false, (b), (short)0, (c), false, false)

__device__ __forceinline__ unsigned pack_bf16(float lo, float hi) {
  union { __bf16 h[2]; unsigned u; } p;
  p.h[0] = (__bf16)lo;                       // native v_cvt bf16 path (RNE)
  p.h[1] = (__bf16)hi;
  return p.u;
}

// softplus(x) - ln2, in base-2 arithmetic on the native transcendentals:
//   softplus(x) = max(x,0) + ln2 * log2(1 + exp2(-|x|*log2e))
//   ssp(x)      = max(x,0) + ln2 * (log2(1 + exp2(-|x|*log2e)) - 1)
__device__ __forceinline__ float sspf(float x) {
  float t = __builtin_amdgcn_exp2f(-fabsf(x) * 1.44269504088896340736f);
  float l = __builtin_amdgcn_logf(1.0f + t);            // log2
  return fmaxf(x, 0.0f) + 0.69314718055994530942f * (l - 1.0f);
}
__device__ __forceinline__ v8f splat8(float b) {
  v8f c = {b, b, b, b, b, b, b, b};
  return c;
}

// A-fragment (16x32 bf16) from a per-lane f32 row pointer.
// Lane l: M = l&15, h = l>>4; VGPR v: K = base32 + (v%4)*2 + 16*(v/4) + 8*h (pair).
__device__ __forceinline__ Frag load_a_f32(const float* __restrict__ row, int base32, int h) {
  Frag f;
#pragma unroll
  for (int v = 0; v < 8; ++v) {
    int k = base32 + ((v & 3) << 1) + ((v >> 2) << 4) + (h << 3);
    float2 d = *reinterpret_cast<const float2*>(row + k);
    f.u[v] = pack_bf16(d.x, d.y);
  }
  return f;
}

// B-fragment (32x16 bf16): lane: N = nbase + (lane&15), h = lane>>4; VGPR v: K = kbase + (v%4)*2 + 16*(v/4) + 8*h (pair).
__device__ __forceinline__ Frag load_b_lds(const unsigned* w, int pitch, int kbase, int nbase, int lane) {
  Frag f;
  const unsigned short* q = reinterpret_cast<const unsigned short*>(w);
  const int n = nbase + (lane & 15), hh = lane >> 4;
#pragma unroll
  for (int v = 0; v < 8; ++v) {
    int k = kbase + ((v & 3) << 1) + ((v >> 2) << 4) + (hh << 3);
    unsigned lo = q[(k * pitch + (n >> 1)) * 2 + (n & 1)];
    unsigned hi = q[((k + 1) * pitch + (n >> 1)) * 2 + (n & 1)];
    f.u[v] = lo | (hi << 16);
  }
  return f;
}

// A-fragment from per-wave staging buffer (pre-swizzled word order).
__device__ __forceinline__ Frag load_stage(const unsigned* stg, int kt, int lane) {
  Frag f;
#pragma unroll
  for (int v = 0; v < 8; ++v) f.u[v] = stg[(kt * 8 + v) * 32 + lane];
  return f;
}

// Convert a C fragment (VGPR r: row=r+8h, lane: col) to A-fragment word order
// in the staging buffer. shfl_xor(1) pairs adjacent columns so even lanes emit
// one packed-bf16 dword store per row.
__device__ __forceinline__ void stage_c(unsigned* stg, v8f c, int col, int h, int l16, bool do_ssp) {
  float vals[8];
#pragma unroll
  for (int r = 0; r < 8; ++r) { float x = c[r]; vals[r] = do_ssp ? sspf(x) : x; }
  int kk = col & 31, kt = col >> 5;
  int v  = ((kk & 7) >> 1) + ((kk >> 4) << 2);
  int h2 = (kk >> 3) & 1;
  int base = (kt * 8 + v) * 32 + 16 * h2 + 8 * h;
  bool writer = ((l16 & 1) == 0);
#pragma unroll
  for (int r = 0; r < 8; ++r) {
    float other = __shfl_xor(vals[r], 1, 32);
    if (writer) stg[base + r] = pack_bf16(vals[r], other);
  }
}

// Cooperative fp32 -> packed-bf16 weight staging into LDS ([K][pitch] dwords).
__device__ __forceinline__ void load_w_lds(unsigned* dst, const float* __restrict__ w,
                                           int K, int Nw, int pitch, int tid, int nthr) {
  int half = Nw >> 1;
  for (int i = tid; i < K * half; i += nthr) {
    int k = i / half, j = i - k * half;
    float2 d = *reinterpret_cast<const float2*>(w + (size_t)k * Nw + 2 * j);
    dst[k * pitch + j] = pack_bf16(d.x, d.y);
  }
}

// ---------------------------------------------------------------------------
// Edge pipeline: new_edge = ssp([x_src|x_dst|rbf] W1 + b1) W2 + b2
//                he       = ssp(new_edge Pw1 + Pb1) Pw2 + Pb2
//                agg[dst] += he   (atomic)
// One wave owns two 16-edge tiles; each B fragment is shared by both.
// ---------------------------------------------------------------------------
__global__ __launch_bounds__(128) void nmp_edge_kernel(
    const float* __restrict__ node_feats, const float* __restrict__ edge_feats,
    const int* __restrict__ src, const int* __restrict__ dst,
    const float* __restrict__ eu_w1, const float* __restrict__ eu_b1,
    const float* __restrict__ eu_w2, const float* __restrict__ eu_b2,
    const float* __restrict__ pe_w1, const float* __restrict__ pe_b1,
    const float* __restrict__ pe_w2, const float* __restrict__ pe_b2,
    float* __restrict__ edge_out, float* __restrict__ agg,
    int E, int numGroups)
{
  __shared__ unsigned sW1[KIN * P1];
  __shared__ unsigned sW2[HID * P2];
  __shared__ unsigned sP1[DIMF * P2];
  __shared__ unsigned sP2w[DIMF * P2];
  __shared__ float sB1[HID], sB2[RBFF], sPB1[DIMF], sPB2[DIMF];
  __shared__ unsigned sStage[EW * 2048];

  const int tid = threadIdx.x, nthr = blockDim.x;
  load_w_lds(sW1, eu_w1, KIN, HID, P1, tid, nthr);
  load_w_lds(sW2, eu_w2, HID, RBFF, P2, tid, nthr);
  load_w_lds(sP1, pe_w1, RBFF, DIMF, P2, tid, nthr);
  load_w_lds(sP2w, pe_w2, DIMF, DIMF, P2, tid, nthr);
  for (int i = tid; i < HID;  i += nthr) sB1[i] = eu_b1[i];
  for (int i = tid; i < RBFF; i += nthr) sB2[i] = eu_b2[i];
  for (int i = tid; i < DIMF; i += nthr) { sPB1[i] = pe_b1[i]; sPB2[i] = pe_b2[i]; }
  __syncthreads();

  const int wave = tid >> 5, lane = tid & 31, l16 = lane & 15, h = lane >> 4;
  unsigned* stg = sStage + wave * 2048;          // two 1024-word tile regions

  for (int g = blockIdx.x; g < numGroups; g += gridDim.x) {
    const int e0 = (g * EW + wave) * 32;         // 32 edges per wave
    if (e0 >= E) continue;                       // wave-uniform

    // ---- GEMM1: [2x 16x192] x [192x128] -> ssp -> stage ----
    Frag a1[2][6];
#pragma unroll
    for (int t = 0; t < 2; ++t) {
      int eL = e0 + t * 16 + l16; if (eL > E - 1) eL = E - 1;
      const float* rowS = node_feats + (size_t)src[eL] * DIMF;
      const float* rowD = node_feats + (size_t)dst[eL] * DIMF;
      const float* rowE = edge_feats + (size_t)eL * RBFF;
      a1[t][0] = load_a_f32(rowS, 0, h);  a1[t][1] = load_a_f32(rowS, 32, h);
      a1[t][2] = load_a_f32(rowD, 0, h);  a1[t][3] = load_a_f32(rowD, 32, h);
      a1[t][4] = load_a_f32(rowE, 0, h);  a1[t][5] = load_a_f32(rowE, 32, h);
    }
#pragma unroll
    for (int nt = 0; nt < 8; ++nt) {
      v8f cc[2];
      cc[0] = splat8(sB1[nt * 16 + l16]);
      cc[1] = cc[0];
#pragma unroll
      for (int kt = 0; kt < 6; ++kt) {
        Frag b = load_b_lds(sW1, P1, kt * 32, nt * 16, lane);
        cc[0] = WMMA_BF16(a1[0][kt].bf, b.bf, cc[0]);
        cc[1] = WMMA_BF16(a1[1][kt].bf, b.bf, cc[1]);
      }
      stage_c(stg,        cc[0], nt * 16 + l16, h, l16, true);
      stage_c(stg + 1024, cc[1], nt * 16 + l16, h, l16, true);
    }

    // ---- GEMM2: [2x 16x128] x [128x64] -> new_edge (store) + stage raw ----
    Frag a2[2][4];
#pragma unroll
    for (int t = 0; t < 2; ++t)
#pragma unroll
      for (int kt = 0; kt < 4; ++kt) a2[t][kt] = load_stage(stg + t * 1024, kt, lane);
#pragma unroll
    for (int nt = 0; nt < 4; ++nt) {
      v8f cc[2];
      cc[0] = splat8(sB2[nt * 16 + l16]);
      cc[1] = cc[0];
#pragma unroll
      for (int kt = 0; kt < 4; ++kt) {
        Frag b = load_b_lds(sW2, P2, kt * 32, nt * 16, lane);
        cc[0] = WMMA_BF16(a2[0][kt].bf, b.bf, cc[0]);
        cc[1] = WMMA_BF16(a2[1][kt].bf, b.bf, cc[1]);
      }
      const int col = nt * 16 + l16;
#pragma unroll
      for (int t = 0; t < 2; ++t) {
#pragma unroll
        for (int r = 0; r < 8; ++r) {
          int e = e0 + t * 16 + r + 8 * h;
          if (e < E) edge_out[(size_t)e * RBFF + col] = cc[t][r];
        }
        stage_c(stg + t * 1024, cc[t], col, h, l16, false);
      }
    }

    // ---- GEMM3: new_edge x Pw1 -> ssp -> stage ----
    Frag a3[2][2];
#pragma unroll
    for (int t = 0; t < 2; ++t)
#pragma unroll
      for (int kt = 0; kt < 2; ++kt) a3[t][kt] = load_stage(stg + t * 1024, kt, lane);
#pragma unroll
    for (int nt = 0; nt < 4; ++nt) {
      v8f cc[2];
      cc[0] = splat8(sPB1[nt * 16 + l16]);
      cc[1] = cc[0];
#pragma unroll
      for (int kt = 0; kt < 2; ++kt) {
        Frag b = load_b_lds(sP1, P2, kt * 32, nt * 16, lane);
        cc[0] = WMMA_BF16(a3[0][kt].bf, b.bf, cc[0]);
        cc[1] = WMMA_BF16(a3[1][kt].bf, b.bf, cc[1]);
      }
      stage_c(stg,        cc[0], nt * 16 + l16, h, l16, true);
      stage_c(stg + 1024, cc[1], nt * 16 + l16, h, l16, true);
    }

    // ---- GEMM4: -> he, scatter-add into agg[dst] ----
    Frag a4[2][2];
#pragma unroll
    for (int t = 0; t < 2; ++t)
#pragma unroll
      for (int kt = 0; kt < 2; ++kt) a4[t][kt] = load_stage(stg + t * 1024, kt, lane);
    int dsts[2][8];
#pragma unroll
    for (int t = 0; t < 2; ++t)
#pragma unroll
      for (int r = 0; r < 8; ++r) {
        int e = e0 + t * 16 + r + 8 * h; if (e > E - 1) e = E - 1;
        dsts[t][r] = dst[e];
      }
#pragma unroll
    for (int nt = 0; nt < 4; ++nt) {
      v8f cc[2];
      cc[0] = splat8(sPB2[nt * 16 + l16]);
      cc[1] = cc[0];
#pragma unroll
      for (int kt = 0; kt < 2; ++kt) {
        Frag b = load_b_lds(sP2w, P2, kt * 32, nt * 16, lane);
        cc[0] = WMMA_BF16(a4[0][kt].bf, b.bf, cc[0]);
        cc[1] = WMMA_BF16(a4[1][kt].bf, b.bf, cc[1]);
      }
      const int col = nt * 16 + l16;
#pragma unroll
      for (int t = 0; t < 2; ++t)
#pragma unroll
        for (int r = 0; r < 8; ++r) {
          int e = e0 + t * 16 + r + 8 * h;
          if (e < E) atomicAdd(&agg[(size_t)dsts[t][r] * DIMF + col], cc[t][r]);
        }
    }
  }
}

// ---------------------------------------------------------------------------
// Node pipeline: out = node_feats + ssp(agg W1 + b1) W2 + b2
// ---------------------------------------------------------------------------
__global__ __launch_bounds__(256) void nmp_node_kernel(
    const float* __restrict__ node_feats, const float* __restrict__ agg,
    const float* __restrict__ w1, const float* __restrict__ b1,
    const float* __restrict__ w2, const float* __restrict__ b2,
    float* __restrict__ out, int N)
{
  __shared__ unsigned sW1[DIMF * P2];
  __shared__ unsigned sW2[DIMF * P2];
  __shared__ float sB1[DIMF], sB2[DIMF];
  __shared__ unsigned sStage[NW * 512];

  const int tid = threadIdx.x, nthr = blockDim.x;
  load_w_lds(sW1, w1, DIMF, DIMF, P2, tid, nthr);
  load_w_lds(sW2, w2, DIMF, DIMF, P2, tid, nthr);
  for (int i = tid; i < DIMF; i += nthr) { sB1[i] = b1[i]; sB2[i] = b2[i]; }
  __syncthreads();

  const int wave = tid >> 5, lane = tid & 31, l16 = lane & 15, h = lane >> 4;
  unsigned* stg = sStage + wave * 512;

  const int n0 = (blockIdx.x * NW + wave) * 16;
  if (n0 >= N) return;
  int nL = n0 + l16; if (nL > N - 1) nL = N - 1;
  const float* row = agg + (size_t)nL * DIMF;
  Frag a[2] = { load_a_f32(row, 0, h), load_a_f32(row, 32, h) };

#pragma unroll
  for (int nt = 0; nt < 4; ++nt) {
    v8f c = splat8(sB1[nt * 16 + l16]);
#pragma unroll
    for (int kt = 0; kt < 2; ++kt) {
      Frag b = load_b_lds(sW1, P2, kt * 32, nt * 16, lane);
      c = WMMA_BF16(a[kt].bf, b.bf, c);
    }
    stage_c(stg, c, nt * 16 + l16, h, l16, true);
  }
  Frag a2f[2];
#pragma unroll
  for (int kt = 0; kt < 2; ++kt) a2f[kt] = load_stage(stg, kt, lane);
#pragma unroll
  for (int nt = 0; nt < 4; ++nt) {
    v8f c = splat8(sB2[nt * 16 + l16]);
#pragma unroll
    for (int kt = 0; kt < 2; ++kt) {
      Frag b = load_b_lds(sW2, P2, kt * 32, nt * 16, lane);
      c = WMMA_BF16(a2f[kt].bf, b.bf, c);
    }
    const int col = nt * 16 + l16;
#pragma unroll
    for (int r = 0; r < 8; ++r) {
      int n = n0 + r + 8 * h;
      if (n < N) out[(size_t)n * DIMF + col] = c[r] + node_feats[(size_t)n * DIMF + col];
    }
  }
}

__global__ void nmp_zero_kernel(float* __restrict__ p, long long n) {
  long long i = (long long)blockIdx.x * blockDim.x + threadIdx.x;
  long long stride = (long long)gridDim.x * blockDim.x;
  for (; i < n; i += stride) p[i] = 0.0f;
}

extern "C" void kernel_launch(void* const* d_in, const int* in_sizes, int n_in,
                              void* d_out, int out_size, void* d_ws, size_t ws_size,
                              hipStream_t stream) {
  const float* node_feats = (const float*)d_in[0];
  const float* edge_feats = (const float*)d_in[1];
  const int*   src        = (const int*)d_in[2];
  const int*   dst        = (const int*)d_in[3];
  const float* eu_w1 = (const float*)d_in[4];
  const float* eu_b1 = (const float*)d_in[5];
  const float* eu_w2 = (const float*)d_in[6];
  const float* eu_b2 = (const float*)d_in[7];
  // d_in[8], d_in[9] = pn1_w / pn1_b: computed-but-unused in the reference.
  const float* pe_w1 = (const float*)d_in[10];
  const float* pe_b1 = (const float*)d_in[11];
  const float* pe_w2 = (const float*)d_in[12];
  const float* pe_b2 = (const float*)d_in[13];
  const float* pn2_w1 = (const float*)d_in[14];
  const float* pn2_b1 = (const float*)d_in[15];
  const float* pn2_w2 = (const float*)d_in[16];
  const float* pn2_b2 = (const float*)d_in[17];

  const int N = in_sizes[0] / DIMF;
  const int E = in_sizes[2];

  float* out_node = (float*)d_out;                       // [N, dim]
  float* out_edge = out_node + (size_t)N * DIMF;         // [E, rbf]
  float* agg      = (float*)d_ws;                        // [N, dim] scratch

  nmp_zero_kernel<<<2048, 256, 0, stream>>>(agg, (long long)N * DIMF);

  const int numGroups = (E + (EW * 32 - 1)) / (EW * 32); // 128 edges per group
  const int blocks    = numGroups < 3072 ? numGroups : 3072;
  nmp_edge_kernel<<<blocks, 128, 0, stream>>>(
      node_feats, edge_feats, src, dst,
      eu_w1, eu_b1, eu_w2, eu_b2,
      pe_w1, pe_b1, pe_w2, pe_b2,
      out_edge, agg, E, numGroups);

  const int nTiles  = (N + 15) / 16;
  const int nBlocks = (nTiles + NW - 1) / NW;
  nmp_node_kernel<<<nBlocks, 256, 0, stream>>>(
      node_feats, agg, pn2_w1, pn2_b1, pn2_w2, pn2_b2, out_node, N);
}
